// QuantLinear_32238024523923
// MI455X (gfx1250) — compile-verified, matrix operand order checked by reading, not run on hardware
//
#include <hip/hip_runtime.h>

typedef __attribute__((ext_vector_type(16))) _Float16 v16h;
typedef __attribute__((ext_vector_type(8)))  _Float16 v8h;
typedef __attribute__((ext_vector_type(8)))  float    v8f;

#define BM 128
#define BN 128
#define BK 64
#define ASTRF 68                       // fp32 per A-LDS row: 64 + 4 pad = 272 B (16B mult, bank-tiling)
#define BSTR  72                       // f16 per B-LDS row: 64 + 8 pad = 144 B (16B mult)
#define ABUF_BYTES (BM * ASTRF * 4)    // 34816
#define BBUF_BYTES (BN * BSTR * 2)     // 18432
#define LDS_BYTES  (2 * ABUF_BYTES + 2 * BBUF_BYTES)  // 106496

// LDS byte offset of a generic pointer into shared memory.
__device__ __forceinline__ unsigned lds_off(const void* p) {
    return (unsigned)(unsigned long long)
        (__attribute__((address_space(3))) const char*)p;
}

// Async copy of the BM x BK fp32 x-tile into LDS (raw fp32; cvt at fragment build).
__device__ __forceinline__ void issue_a_async(const float* __restrict__ x,
                                              int m0, int K, int k0,
                                              unsigned abase, int tid) {
    #pragma unroll
    for (int i = 0; i < 8; ++i) {
        int idx = tid + i * 256;            // 0..2047 float4 units (128 rows x 16)
        int m   = idx >> 4;
        int c4  = idx & 15;
        const float* g = x + (size_t)(m0 + m) * K + k0 + c4 * 4;
        unsigned     l = abase + (unsigned)(m * ASTRF + c4 * 4) * 4u;
        asm volatile("global_load_async_to_lds_b128 %0, %1, off"
                     :: "v"(l), "v"(g) : "memory");
    }
}

struct BStage {                 // next-stage quant data staged in registers
    unsigned qw[4];
    unsigned qz[4];
    float    qs[4], z0[4], s0[4];
};

__device__ __forceinline__ void load_b_stage(BStage& st,
        const int* __restrict__ qweight, const int* __restrict__ qzeros,
        const int* __restrict__ qscales, const float* __restrict__ qszero,
        const float* __restrict__ qsscale,
        int N, int k0, int n0, int gshift, int tid) {
    #pragma unroll
    for (int i = 0; i < 4; ++i) {
        int idx = tid + i * 256;            // 8 packed rows x 128 cols
        int qk  = idx >> 7;
        int n   = idx & 127;
        int g   = (k0 + qk * 8) >> gshift;  // all 8 nibbles share one group (G >= 8)
        int gn  = n0 + n;
        st.qw[i] = (unsigned)qweight[(size_t)((k0 >> 3) + qk) * N + gn];
        st.qz[i] = (unsigned)qzeros[(size_t)g * (N >> 3) + (gn >> 3)];
        st.qs[i] = (float)qscales[(size_t)g * N + gn];
        st.z0[i] = qszero[g];
        st.s0[i] = qsscale[g];
    }
}

__device__ __forceinline__ void dequant_store(const BStage& st,
                                              _Float16* __restrict__ Bh, int tid) {
    #pragma unroll
    for (int i = 0; i < 4; ++i) {
        int idx = tid + i * 256;
        int qk  = idx >> 7;
        int n   = idx & 127;                // n0 % 8 == 0 -> (n0+n)&7 == n&7
        int   z     = (int)((st.qz[i] >> ((n & 7) * 4)) & 15u) + 1;
        float scale = (st.qs[i] - st.z0[i]) * st.s0[i];
        float zs    = (float)z * scale;
        v8h wv;
        #pragma unroll
        for (int j = 0; j < 8; ++j) {
            int w = (int)((st.qw[i] >> (4 * j)) & 15u);
            wv[j] = (_Float16)((float)w * scale - zs);
        }
        *(v8h*)(&Bh[n * BSTR + qk * 8]) = wv;
    }
}

// Build one 16x32 f16 A fragment from the fp32 LDS tile (layout per ISA 7.12.2).
__device__ __forceinline__ v16h make_afrag(const float* __restrict__ p) {
    float4 q0 = *(const float4*)(p);
    float4 q1 = *(const float4*)(p + 4);
    float4 q2 = *(const float4*)(p + 16);
    float4 q3 = *(const float4*)(p + 20);
    v16h a;
    a[0]=(_Float16)q0.x;  a[1]=(_Float16)q0.y;  a[2]=(_Float16)q0.z;  a[3]=(_Float16)q0.w;
    a[4]=(_Float16)q1.x;  a[5]=(_Float16)q1.y;  a[6]=(_Float16)q1.z;  a[7]=(_Float16)q1.w;
    a[8]=(_Float16)q2.x;  a[9]=(_Float16)q2.y;  a[10]=(_Float16)q2.z; a[11]=(_Float16)q2.w;
    a[12]=(_Float16)q3.x; a[13]=(_Float16)q3.y; a[14]=(_Float16)q3.z; a[15]=(_Float16)q3.w;
    return a;
}

__global__ __launch_bounds__(256)
void w4a16_gemm_wmma(const float* __restrict__ x,
                     const int*   __restrict__ qweight,   // (K/8, N)
                     const int*   __restrict__ qzeros,    // (groups, N/8)
                     const int*   __restrict__ qscales,   // (groups, N)
                     const float* __restrict__ qszero,    // (groups,)
                     const float* __restrict__ qsscale,   // (groups,)
                     float*       __restrict__ out,
                     int M, int N, int K, int gshift)
{
    extern __shared__ __align__(16) char smem[];
    // layout: [A buf0 | A buf1 | B buf0 | B buf1]
    float*    Abase = (float*)smem;
    _Float16* Bbase = (_Float16*)(smem + 2 * ABUF_BYTES);

    const int tid  = threadIdx.x;
    const int lane = tid & 31;
    const int wave = tid >> 5;
    const int wm   = wave & 3;          // 4 waves along M (32 rows each)
    const int wn   = wave >> 2;         // 2 waves along N (64 cols each)
    const int l16  = lane & 15;
    const int hK   = (lane >> 4) * 8;   // 16-bit WMMA operand K-chunk select

    const int m0 = blockIdx.y * BM;
    const int n0 = blockIdx.x * BN;

    const unsigned a_off = lds_off(Abase);

    v8f acc[2][4] = {};

    BStage st;
    issue_a_async(x, m0, K, /*k0=*/0, a_off, tid);
    load_b_stage(st, qweight, qzeros, qscales, qszero, qsscale, N, 0, n0, gshift, tid);

    int buf = 0;
    for (int k0 = 0; k0 < K; k0 += BK, buf ^= 1) {
        _Float16* Bh = Bbase + (size_t)buf * (BN * BSTR);
        dequant_store(st, Bh, tid);

        asm volatile("s_wait_asynccnt 0x0" ::: "memory");  // our A[buf] copies done
        __syncthreads();                                   // everyone's A/B[buf] visible

        if (k0 + BK < K) {                                 // overlap next stage with compute
            issue_a_async(x, m0, K, k0 + BK,
                          a_off + (unsigned)((buf ^ 1) * ABUF_BYTES), tid);
            load_b_stage(st, qweight, qzeros, qscales, qszero, qsscale,
                         N, k0 + BK, n0, gshift, tid);
        }

        const float* Af = Abase + (size_t)buf * (BM * ASTRF);

        #pragma unroll
        for (int ks = 0; ks < BK; ks += 32) {
            v16h afrag[2], bfrag[4];
            #pragma unroll
            for (int i = 0; i < 2; ++i)
                afrag[i] = make_afrag(Af + (wm * 32 + i * 16 + l16) * ASTRF + ks + hK);
            #pragma unroll
            for (int j = 0; j < 4; ++j) {
                const _Float16* p = &Bh[(wn * 64 + j * 16 + l16) * BSTR + ks + hK];
                v8h lo = *(const v8h*)p;
                v8h hi = *(const v8h*)(p + 16);
                bfrag[j] = __builtin_shufflevector(lo, hi,
                    0,1,2,3,4,5,6,7,8,9,10,11,12,13,14,15);
            }
            #pragma unroll
            for (int i = 0; i < 2; ++i)
                #pragma unroll
                for (int j = 0; j < 4; ++j)
                    acc[i][j] = __builtin_amdgcn_wmma_f32_16x16x32_f16(
                        false, afrag[i], false, bfrag[j],
                        (short)0, acc[i][j], false, false);
        }
        // no trailing barrier needed: next iteration's writes only touch buf^1,
        // whose last readers finished before this iteration's barrier.
    }

    // epilogue: C/D VGPR layout -> global f32
    #pragma unroll
    for (int i = 0; i < 2; ++i) {
        int mbase = m0 + wm * 32 + i * 16 + (lane >> 4) * 8;
        #pragma unroll
        for (int j = 0; j < 4; ++j) {
            int nn = n0 + wn * 64 + j * 16 + l16;
            #pragma unroll
            for (int r = 0; r < 8; ++r)
                out[(size_t)(mbase + r) * N + nn] = acc[i][j][r];
        }
    }
}

extern "C" void kernel_launch(void* const* d_in, const int* in_sizes, int n_in,
                              void* d_out, int out_size, void* d_ws, size_t ws_size,
                              hipStream_t stream) {
    const float* x        = (const float*)d_in[0];
    const int*   qweight  = (const int*)  d_in[1];
    const int*   qzeros   = (const int*)  d_in[2];
    const int*   qscales  = (const int*)  d_in[3];
    const float* qszero   = (const float*)d_in[4];
    const float* qsscale  = (const float*)d_in[5];
    // d_in[6] = g_idx: arange(K)/G -> contiguous groups exploited in-kernel.

    const int K      = in_sizes[6];
    const int M      = (int)((long long)in_sizes[0] / K);
    const int N      = (int)((long long)in_sizes[1] * 8 / K);
    const int groups = in_sizes[4];
    const int G      = K / groups;
    const int gshift = __builtin_ctz((unsigned)G);

    dim3 grid((unsigned)(N / BN), (unsigned)(M / BM));
    w4a16_gemm_wmma<<<grid, 256, LDS_BYTES, stream>>>(x, qweight, qzeros, qscales,
                                                      qszero, qsscale, (float*)d_out,
                                                      M, N, K, gshift);
}
